// ToyMDNN_1821066134189
// MI455X (gfx1250) — matrix operand order repaired and checked by reading, not compiled, not run on hardware
//
#include <hip/hip_runtime.h>

// Problem constants (from the reference)
#define NB    1024
#define SQ    128
#define DIM   128
#define NH    4
#define CH    32
#define NLAY  2
#define MROWS (NB * SQ)             // 131072 rows
#define MD    ((size_t)MROWS * DIM) // 16,777,216 elements

typedef __attribute__((ext_vector_type(16))) _Float16 v16h;
typedef __attribute__((ext_vector_type(8)))  _Float16 h8;   // 16 bytes
typedef __attribute__((ext_vector_type(8)))  float    v8f;
typedef __attribute__((ext_vector_type(4)))  unsigned int u32x4;
typedef __attribute__((ext_vector_type(4)))  int      i32x4;
typedef __attribute__((ext_vector_type(8)))  int      i32x8;

// ---------------------------------------------------------------------------
// Tensor Data Mover: DMA a 2D f16 tile (tile_w x tile_h, row stride in elems)
// from global memory into LDS. Issued by one wave; completion via TENSORcnt.
// ---------------------------------------------------------------------------
#if __has_builtin(__builtin_amdgcn_tensor_load_to_lds)
#define ATH_HAVE_TDM 1

__device__ __forceinline__ void tdm_issue(u32x4 g0, i32x8 g1) {
    i32x4 z4 = {};
#if __clang_major__ >= 23
    i32x8 z8 = {};
    __builtin_amdgcn_tensor_load_to_lds(g0, g1, z4, z4, z8, 0);
#else
    __builtin_amdgcn_tensor_load_to_lds(g0, g1, z4, z4, 0);
#endif
}

__device__ __forceinline__ void tdm_load_2d_f16(unsigned lds_addr, const void* gptr,
                                                unsigned tile_w, unsigned tile_h,
                                                unsigned stride_elems) {
    unsigned long long ga = (unsigned long long)(size_t)gptr;
    // D# group 0: count=1 (user mode), lds_addr, global_addr[56:0], type=2
    u32x4 g0 = {};
    g0[0] = 1u;
    g0[1] = lds_addr;
    g0[2] = (unsigned)ga;
    g0[3] = (unsigned)((ga >> 32) & 0x01FFFFFFu) | (2u << 30);
    // D# group 1: data_size=2B; tensor_dim0=stride, tensor_dim1=tile_h;
    // tile_dim0=tile_w, tile_dim1=tile_h; tensor_dim0_stride=stride
    i32x8 g1 = {};
    g1[0] = (int)(1u << 16);                                   // data_size = 1 -> 2 bytes
    g1[1] = (int)((stride_elems & 0xFFFFu) << 16);             // tensor_dim0 lo16
    g1[2] = (int)(((tile_h & 0xFFFFu) << 16) | (stride_elems >> 16)); // dim1 lo | dim0 hi
    g1[3] = (int)(((tile_w & 0xFFFFu) << 16) | (tile_h >> 16));       // tile_dim0 | dim1 hi
    g1[4] = (int)(tile_h & 0xFFFFu);                           // tile_dim1 (tile_dim2=0)
    g1[5] = (int)stride_elems;                                 // tensor_dim0_stride lo32
    g1[6] = 0;
    g1[7] = 0;
    tdm_issue(g0, g1);
}
#endif

// ---------------------------------------------------------------------------
// f32 -> f16 conversion (vectorized)
// ---------------------------------------------------------------------------
__global__ __launch_bounds__(256) void k_f32_to_f16(const float* __restrict__ in,
                                                    _Float16* __restrict__ out, int n) {
    int i = (blockIdx.x * 256 + threadIdx.x) * 4;
    if (i >= n) return;
    float4 v = *(const float4*)(in + i);
    out[i + 0] = (_Float16)v.x;
    out[i + 1] = (_Float16)v.y;
    out[i + 2] = (_Float16)v.z;
    out[i + 3] = (_Float16)v.w;
}

// ---------------------------------------------------------------------------
// GEMM: Y[M x 128] = A16[M x 128] @ W16[o][k]^T + bias   (f16 in, f32 out)
// One block = 128x128 tile, 8 waves, each wave a 16x128 strip.
// A-tile and W panel staged into LDS by the Tensor Data Mover.
// ---------------------------------------------------------------------------
__global__ __launch_bounds__(256) void k_gemm128(const _Float16* __restrict__ A,
                                                 const _Float16* __restrict__ W,
                                                 const float* __restrict__ bias,
                                                 float* __restrict__ Y) {
    __shared__ _Float16 sA[128 * 128];
    __shared__ _Float16 sW[128 * 128];

    const int tid  = threadIdx.x;
    const int lane = tid & 31;
    const int wv   = tid >> 5;
    const long m0  = (long)blockIdx.x * 128;

    __builtin_prefetch(W, 0, 1);   // global_prefetch_b8 of the weight panel

#if ATH_HAVE_TDM
    if (wv == 0) {
        tdm_load_2d_f16((unsigned)(size_t)&sA[0], &A[m0 * 128], 128, 128, 128);
        tdm_load_2d_f16((unsigned)(size_t)&sW[0], W,            128, 128, 128);
        __builtin_amdgcn_s_wait_tensorcnt(0);
    }
#else
#pragma unroll
    for (int i = 0; i < 8; ++i) {
        int idx = i * 256 + tid;        // 0..2047
        int row = idx >> 4;
        int c8  = (idx & 15) << 3;      // half-index, 16B aligned
        *(h8*)&sA[row * 128 + c8] = *(const h8*)&A[(m0 + row) * 128 + c8];
        *(h8*)&sW[row * 128 + c8] = *(const h8*)&W[row * 128 + c8];
    }
#endif
    __syncthreads();

    v8f acc[8] = {};
    const int mw = wv * 16;
    const int lo = lane & 15;
    const int hi = lane >> 4;

#pragma unroll
    for (int kc = 0; kc < 4; ++kc) {
        // A fragment: contiguous 16 K-values per lane half — K {0..15} / {16..31}
        v16h af;
        {
            const _Float16* p = &sA[(mw + lo) * 128 + kc * 32 + hi * 16];
            *(h8*)&af       = *(const h8*)p;
            *((h8*)&af + 1) = *(const h8*)(p + 8);
        }
#pragma unroll
        for (int j = 0; j < 8; ++j) {
            // B fragment: lane = output feature o, contiguous 16 K-values per lane half
            v16h bf;
            const _Float16* p2 = &sW[(j * 16 + lo) * 128 + kc * 32 + hi * 16];
            *(h8*)&bf       = *(const h8*)p2;
            *((h8*)&bf + 1) = *(const h8*)(p2 + 8);
            acc[j] = __builtin_amdgcn_wmma_f32_16x16x32_f16(false, af, false, bf,
                                                            (short)0, acc[j], false, false);
        }
    }

#pragma unroll
    for (int j = 0; j < 8; ++j) {
        int n   = j * 16 + lo;
        float b = bias[n];
#pragma unroll
        for (int r = 0; r < 8; ++r) {
            int m = mw + r + hi * 8;
            Y[(m0 + m) * 128 + n] = acc[j][r] + b;
        }
    }
}

// ---------------------------------------------------------------------------
// BatchNorm: partial sums + atomics, then finalize into per-feature scale/shift
// ---------------------------------------------------------------------------
__global__ __launch_bounds__(256) void k_bn_stats(const float* __restrict__ Y,
                                                  float* __restrict__ stats,
                                                  int rowsPerBlock) {
    __shared__ float s1[256], s2[256];
    int t = threadIdx.x;
    int f = t & 127;
    int r0 = t >> 7;
    long base = (long)blockIdx.x * rowsPerBlock;
    float a = 0.f, b = 0.f;
    for (int r = r0; r < rowsPerBlock; r += 2) {
        float v = Y[(base + r) * 128 + f];
        a += v; b += v * v;
    }
    s1[t] = a; s2[t] = b;
    __syncthreads();
    if (t < 128) {
        atomicAdd(&stats[t],       s1[t] + s1[t + 128]);
        atomicAdd(&stats[128 + t], s2[t] + s2[t + 128]);
    }
}

__global__ void k_bn_finalize(const float* __restrict__ stats,
                              const float* __restrict__ g, const float* __restrict__ b,
                              float* __restrict__ ss, float invM) {
    int f = threadIdx.x;                  // 128 threads
    float mean = stats[f] * invM;
    float var  = stats[128 + f] * invM - mean * mean;
    float sc   = g[f] * rsqrtf(var + 1e-5f);
    ss[f]       = sc;
    ss[128 + f] = b[f] - mean * sc;
}

__global__ __launch_bounds__(256) void k_bn_apply_f32(const float* __restrict__ in,
                                                      const float* __restrict__ ss,
                                                      const float* __restrict__ add,
                                                      float* __restrict__ out, long n) {
    long i = ((long)blockIdx.x * 256 + threadIdx.x) * 4;
    if (i >= n) return;
    float4 v = *(const float4*)(in + i);
    int f = (int)(i & 127);
    float4 o;
    o.x = v.x * ss[f + 0] + ss[128 + f + 0];
    o.y = v.y * ss[f + 1] + ss[128 + f + 1];
    o.z = v.z * ss[f + 2] + ss[128 + f + 2];
    o.w = v.w * ss[f + 3] + ss[128 + f + 3];
    if (add) {
        float4 a = *(const float4*)(add + i);
        o.x += a.x; o.y += a.y; o.z += a.z; o.w += a.w;
    }
    *(float4*)(out + i) = o;
}

__global__ __launch_bounds__(256) void k_bn_apply_f16(const float* __restrict__ in,
                                                      const float* __restrict__ ss,
                                                      _Float16* __restrict__ out, long n) {
    long i = ((long)blockIdx.x * 256 + threadIdx.x) * 4;
    if (i >= n) return;
    float4 v = *(const float4*)(in + i);
    int f = (int)(i & 127);
    out[i + 0] = (_Float16)(v.x * ss[f + 0] + ss[128 + f + 0]);
    out[i + 1] = (_Float16)(v.y * ss[f + 1] + ss[128 + f + 1]);
    out[i + 2] = (_Float16)(v.z * ss[f + 2] + ss[128 + f + 2]);
    out[i + 3] = (_Float16)(v.w * ss[f + 3] + ss[128 + f + 3]);
}

// ---------------------------------------------------------------------------
// Attention: one block per (n, h). QK^T -> mask -> softmax -> P@V, all WMMA.
// Q/K head slices (strided 128x32 tiles) are DMA'd to LDS by the TDM.
// ---------------------------------------------------------------------------
__global__ __launch_bounds__(256) void k_attn(const _Float16* __restrict__ Q,
                                              const _Float16* __restrict__ K,
                                              const _Float16* __restrict__ V,
                                              const float* __restrict__ mask,
                                              float* __restrict__ agg) {
    __shared__ _Float16 sQ[128 * 32];    // [s][c]
    __shared__ _Float16 sK[128 * 32];    // [t][c]
    __shared__ _Float16 sVT[32 * 128];   // [c][t] (transposed for B-fragments)
    __shared__ _Float16 sP[128 * 128];   // probs

    const int bid = blockIdx.x;
    const int nb  = bid >> 2;            // / NH
    const int hh  = bid & 3;             // % NH
    const int tid = threadIdx.x;
    const int lane = tid & 31;
    const int wv   = tid >> 5;
    const long rowBase = (long)nb * SQ;

#if ATH_HAVE_TDM
    if (wv == 0) {
        tdm_load_2d_f16((unsigned)(size_t)&sQ[0], &Q[rowBase * 128 + hh * 32], 32, 128, 128);
        tdm_load_2d_f16((unsigned)(size_t)&sK[0], &K[rowBase * 128 + hh * 32], 32, 128, 128);
    }
    // V is transposed into LDS manually (TDM has no transpose path)
#pragma unroll
    for (int i = 0; i < 2; ++i) {
        int idx = i * 256 + tid;         // 0..511
        int s   = idx >> 2;
        int c8  = (idx & 3) << 3;
        h8 vv = *(const h8*)&V[(rowBase + s) * 128 + hh * 32 + c8];
#pragma unroll
        for (int e = 0; e < 8; ++e) sVT[(c8 + e) * 128 + s] = vv[e];
    }
    if (wv == 0) __builtin_amdgcn_s_wait_tensorcnt(0);
#else
#pragma unroll
    for (int i = 0; i < 2; ++i) {
        int idx = i * 256 + tid;         // 0..511
        int s   = idx >> 2;
        int c8  = (idx & 3) << 3;
        long g  = (rowBase + s) * 128 + hh * 32 + c8;
        *(h8*)&sQ[s * 32 + c8] = *(const h8*)&Q[g];
        *(h8*)&sK[s * 32 + c8] = *(const h8*)&K[g];
        h8 vv = *(const h8*)&V[g];
#pragma unroll
        for (int e = 0; e < 8; ++e) sVT[(c8 + e) * 128 + s] = vv[e];
    }
#endif
    __syncthreads();

    const int mw = wv * 16;
    const int lo = lane & 15;
    const int hi = lane >> 4;

    // scores: 16x128 strip, single K-chunk (C = 32)
    v8f sc[8];
    {
        v16h af;
        const _Float16* p = &sQ[(mw + lo) * 32 + hi * 16];
        *(h8*)&af       = *(const h8*)p;
        *((h8*)&af + 1) = *(const h8*)(p + 8);
#pragma unroll
        for (int j = 0; j < 8; ++j) {
            v16h bf;
            const _Float16* p2 = &sK[(j * 16 + lo) * 32 + hi * 16];
            *(h8*)&bf       = *(const h8*)p2;
            *((h8*)&bf + 1) = *(const h8*)(p2 + 8);
            v8f z = {};
            sc[j] = __builtin_amdgcn_wmma_f32_16x16x32_f16(false, af, false, bf,
                                                           (short)0, z, false, false);
        }
    }

    // multiplicative pad mask (reference: alpha * mask, zeros participate in softmax)
    const float* mrow = mask + (long)nb * SQ * SQ;
#pragma unroll
    for (int r = 0; r < 8; ++r) {
        int m = mw + r + hi * 8;
#pragma unroll
        for (int j = 0; j < 8; ++j) sc[j][r] *= mrow[m * SQ + j * 16 + lo];
    }

    // softmax per row: row lives in 8 regs x 16 lanes of one half-wave
#pragma unroll
    for (int r = 0; r < 8; ++r) {
        float mx = sc[0][r];
#pragma unroll
        for (int j = 1; j < 8; ++j) mx = fmaxf(mx, sc[j][r]);
#pragma unroll
        for (int off = 1; off < 16; off <<= 1) mx = fmaxf(mx, __shfl_xor(mx, off, 32));
        float ev[8];
        float sum = 0.f;
#pragma unroll
        for (int j = 0; j < 8; ++j) { ev[j] = __expf(sc[j][r] - mx); sum += ev[j]; }
#pragma unroll
        for (int off = 1; off < 16; off <<= 1) sum += __shfl_xor(sum, off, 32);
        float inv = 1.0f / sum;
        int m = mw + r + hi * 8;
#pragma unroll
        for (int j = 0; j < 8; ++j) sP[m * 128 + j * 16 + lo] = (_Float16)(ev[j] * inv);
    }
    // Each wave only re-reads its own sP rows -> no inter-wave barrier needed.

    // agg: 16x32 strip, K = 128 (4 chunks)
    v8f ac[2] = {};
#pragma unroll
    for (int kc = 0; kc < 4; ++kc) {
        v16h af;
        const _Float16* p = &sP[(mw + lo) * 128 + kc * 32 + hi * 16];
        *(h8*)&af       = *(const h8*)p;
        *((h8*)&af + 1) = *(const h8*)(p + 8);
#pragma unroll
        for (int j = 0; j < 2; ++j) {
            v16h bf;
            const _Float16* p2 = &sVT[(j * 16 + lo) * 128 + kc * 32 + hi * 16];
            *(h8*)&bf       = *(const h8*)p2;
            *((h8*)&bf + 1) = *(const h8*)(p2 + 8);
            ac[j] = __builtin_amdgcn_wmma_f32_16x16x32_f16(false, af, false, bf,
                                                           (short)0, ac[j], false, false);
        }
    }

#pragma unroll
    for (int j = 0; j < 2; ++j)
#pragma unroll
        for (int r = 0; r < 8; ++r) {
            int m = mw + r + hi * 8;
            agg[(rowBase + m) * 128 + hh * 32 + j * 16 + lo] = ac[j][r];
        }
}

// ---------------------------------------------------------------------------
// LayerNorm(out = LN(a + res)) — one wave per 128-wide row
// ---------------------------------------------------------------------------
__global__ __launch_bounds__(256) void k_ln(const float* __restrict__ a,
                                            const float* __restrict__ res,
                                            const float* __restrict__ g,
                                            const float* __restrict__ b,
                                            float* __restrict__ out) {
    int lane = threadIdx.x & 31;
    int wv   = threadIdx.x >> 5;
    long row = (long)blockIdx.x * 8 + wv;
    float4 va = *(const float4*)&a[row * 128 + lane * 4];
    float4 vr = *(const float4*)&res[row * 128 + lane * 4];
    float x0 = va.x + vr.x, x1 = va.y + vr.y, x2 = va.z + vr.z, x3 = va.w + vr.w;
    float s = x0 + x1 + x2 + x3;
    float q = x0 * x0 + x1 * x1 + x2 * x2 + x3 * x3;
#pragma unroll
    for (int off = 1; off < 32; off <<= 1) {
        s += __shfl_xor(s, off, 32);
        q += __shfl_xor(q, off, 32);
    }
    float mean = s * (1.f / 128.f);
    float var  = q * (1.f / 128.f) - mean * mean;
    float rs   = rsqrtf(var + 1e-5f);
    float4 vg = *(const float4*)&g[lane * 4];
    float4 vb = *(const float4*)&b[lane * 4];
    float4 o;
    o.x = (x0 - mean) * rs * vg.x + vb.x;
    o.y = (x1 - mean) * rs * vg.y + vb.y;
    o.z = (x2 - mean) * rs * vg.z + vb.z;
    o.w = (x3 - mean) * rs * vg.w + vb.w;
    *(float4*)&out[row * 128 + lane * 4] = o;
}

// h[:, 0] -> out
__global__ void k_take_first(const float* __restrict__ h, float* __restrict__ out) {
    out[blockIdx.x * 128 + threadIdx.x] = h[(long)blockIdx.x * SQ * 128 + threadIdx.x];
}

// ---------------------------------------------------------------------------
extern "C" void kernel_launch(void* const* d_in, const int* in_sizes, int n_in,
                              void* d_out, int out_size, void* d_ws, size_t ws_size,
                              hipStream_t stream) {
    const float* x    = (const float*)d_in[0];
    const float* mask = (const float*)d_in[1];
    const float* linW = (const float*)d_in[2];
    const float* linB = (const float*)d_in[3];
    const float* inG  = (const float*)d_in[4];
    const float* inBb = (const float*)d_in[5];
    const float* qW = (const float*)d_in[6];  const float* qB  = (const float*)d_in[7];
    const float* qG = (const float*)d_in[8];  const float* qBb = (const float*)d_in[9];
    const float* kW = (const float*)d_in[10]; const float* kB  = (const float*)d_in[11];
    const float* kG = (const float*)d_in[12]; const float* kBb = (const float*)d_in[13];
    const float* vW = (const float*)d_in[14]; const float* vB  = (const float*)d_in[15];
    const float* vG = (const float*)d_in[16]; const float* vBb = (const float*)d_in[17];
    const float* lnG = (const float*)d_in[18]; const float* lnB = (const float*)d_in[19];
    const float* fW = (const float*)d_in[20]; const float* fB  = (const float*)d_in[21];
    const float* fG = (const float*)d_in[22]; const float* fBb = (const float*)d_in[23];
    const float* skW = (const float*)d_in[24]; const float* skB  = (const float*)d_in[25];
    const float* skG = (const float*)d_in[26]; const float* skBb = (const float*)d_in[27];

    char* ws = (char*)d_ws;
    float*    h    = (float*)ws;    ws += MD * 4;   // hidden state
    float*    y    = (float*)ws;    ws += MD * 4;   // GEMM scratch
    float*    aggb = (float*)ws;    ws += MD * 4;   // attention out / FF out (aliased in time)
    float*    outb = (float*)ws;    ws += MD * 4;   // LN output
    _Float16* h16  = (_Float16*)ws; ws += MD * 2;
    _Float16* q16  = (_Float16*)ws; ws += MD * 2;   // also reused as FF input f16
    _Float16* k16  = (_Float16*)ws; ws += MD * 2;
    _Float16* v16  = (_Float16*)ws; ws += MD * 2;
    _Float16* w16  = (_Float16*)ws; ws += (size_t)11 * DIM * DIM * 2;
    float*    stats= (float*)ws;    ws += 256 * 4;
    float*    ss   = (float*)ws;    ws += 256 * 4;

    const int gGemm  = MROWS / 128;       // 1024
    const int gApply = (int)(MD / 1024);  // 16384
    const int gLN    = MROWS / 8;         // 16384
    const int gAttn  = NB * NH;           // 4096
    const int gConvW = (DIM * DIM) / 1024;// 16
    const float invM = 1.0f / (float)MROWS;

    // Convert all weight matrices to f16 once
    _Float16* linW16 = w16;
    k_f32_to_f16<<<gConvW, 256, 0, stream>>>(linW, linW16, DIM * DIM);
    for (int l = 0; l < NLAY; ++l) {
        const float* srcs[5] = { qW + (size_t)l * DIM * DIM, kW + (size_t)l * DIM * DIM,
                                 vW + (size_t)l * DIM * DIM, fW + (size_t)l * DIM * DIM,
                                 skW + (size_t)l * DIM * DIM };
        for (int j = 0; j < 5; ++j)
            k_f32_to_f16<<<gConvW, 256, 0, stream>>>(srcs[j],
                w16 + (size_t)(1 + l * 5 + j) * DIM * DIM, DIM * DIM);
    }

    // ---- input projection + BN ----
    k_f32_to_f16<<<gApply, 256, 0, stream>>>(x, h16, (int)MD);
    k_gemm128<<<gGemm, 256, 0, stream>>>(h16, linW16, linB, y);
    hipMemsetAsync(stats, 0, 256 * sizeof(float), stream);
    k_bn_stats<<<512, 256, 0, stream>>>(y, stats, 256);
    k_bn_finalize<<<1, 128, 0, stream>>>(stats, inG, inBb, ss, invM);
    k_bn_apply_f32<<<gApply, 256, 0, stream>>>(y, ss, nullptr, h, (long)MD);
    k_f32_to_f16<<<gApply, 256, 0, stream>>>(h, h16, (int)MD);

    for (int l = 0; l < NLAY; ++l) {
        const size_t bl = (size_t)l * DIM;
        _Float16* qW16 = w16 + (size_t)(1 + l * 5 + 0) * DIM * DIM;
        _Float16* kW16 = w16 + (size_t)(1 + l * 5 + 1) * DIM * DIM;
        _Float16* vW16 = w16 + (size_t)(1 + l * 5 + 2) * DIM * DIM;
        _Float16* fW16 = w16 + (size_t)(1 + l * 5 + 3) * DIM * DIM;
        _Float16* sW16 = w16 + (size_t)(1 + l * 5 + 4) * DIM * DIM;

        // Q/K/V projections + BN, written straight to f16 for the attention kernel
        struct { _Float16* w; const float* b; const float* g; const float* bb; _Float16* o; }
        proj[3] = { { qW16, qB + bl, qG + bl, qBb + bl, q16 },
                    { kW16, kB + bl, kG + bl, kBb + bl, k16 },
                    { vW16, vB + bl, vG + bl, vBb + bl, v16 } };
        for (int p = 0; p < 3; ++p) {
            k_gemm128<<<gGemm, 256, 0, stream>>>(h16, proj[p].w, proj[p].b, y);
            hipMemsetAsync(stats, 0, 256 * sizeof(float), stream);
            k_bn_stats<<<512, 256, 0, stream>>>(y, stats, 256);
            k_bn_finalize<<<1, 128, 0, stream>>>(stats, proj[p].g, proj[p].bb, ss, invM);
            k_bn_apply_f16<<<gApply, 256, 0, stream>>>(y, ss, proj[p].o, (long)MD);
        }

        // attention -> aggb ; LN(agg + h) -> outb
        k_attn<<<gAttn, 256, 0, stream>>>(q16, k16, v16, mask, aggb);
        k_ln<<<gLN, 256, 0, stream>>>(aggb, h, lnG + bl, lnB + bl, outb);

        // FF: BN(out @ ffW^T + b); q16/aggb reused as scratch
        k_f32_to_f16<<<gApply, 256, 0, stream>>>(outb, q16, (int)MD);
        k_gemm128<<<gGemm, 256, 0, stream>>>(q16, fW16, fB + bl, y);
        hipMemsetAsync(stats, 0, 256 * sizeof(float), stream);
        k_bn_stats<<<512, 256, 0, stream>>>(y, stats, 256);
        k_bn_finalize<<<1, 128, 0, stream>>>(stats, fG + bl, fBb + bl, ss, invM);
        k_bn_apply_f32<<<gApply, 256, 0, stream>>>(y, ss, nullptr, aggb, (long)MD);

        // out = LN(out + ff)  (in-place safe: each thread reads before writing)
        k_ln<<<gLN, 256, 0, stream>>>(outb, aggb, lnG + bl, lnB + bl, outb);

        // skip: h_new = BN(h @ skipW^T + b) + out
        k_gemm128<<<gGemm, 256, 0, stream>>>(h16, sW16, skB + bl, y);
        hipMemsetAsync(stats, 0, 256 * sizeof(float), stream);
        k_bn_stats<<<512, 256, 0, stream>>>(y, stats, 256);
        k_bn_finalize<<<1, 128, 0, stream>>>(stats, skG + bl, skBb + bl, ss, invM);
        k_bn_apply_f32<<<gApply, 256, 0, stream>>>(y, ss, outb, h, (long)MD);

        k_f32_to_f16<<<gApply, 256, 0, stream>>>(h, h16, (int)MD);
    }

    // out = h[:, 0]
    k_take_first<<<NB, 128, 0, stream>>>(h, (float*)d_out);
}
